// LePEAttention_704374636881
// MI455X (gfx1250) — matrix operand order repaired and checked by reading, not compiled, not run on hardware
//
#include <hip/hip_runtime.h>
#include <hip/hip_bf16.h>

#define BATCH 32
#define SEQ   1024
#define DIM   1024

typedef __attribute__((ext_vector_type(16))) __bf16 bf16x16;
typedef __attribute__((ext_vector_type(8)))  __bf16 bf16x8;
typedef __attribute__((ext_vector_type(4)))  __bf16 bf16x4;
typedef __attribute__((ext_vector_type(8)))  float  f32x8;
typedef __attribute__((ext_vector_type(4)))  float  fvec4;

__device__ inline f32x8 wmma_bf16(bf16x16 a, bf16x16 b, f32x8 c) {
  return __builtin_amdgcn_wmma_f32_16x16x32_bf16(false, a, false, b, (short)0, c, false, false);
}

__device__ inline f32x8 zero8() {
  f32x8 z;
#pragma unroll
  for (int i = 0; i < 8; ++i) z[i] = 0.f;
  return z;
}

__device__ inline bf16x16 cat16(bf16x8 a, bf16x8 b) {
  return __builtin_shufflevector(a, b, 0, 1, 2, 3, 4, 5, 6, 7, 8, 9, 10, 11, 12, 13, 14, 15);
}

__device__ inline bf16x8 cvt8(fvec4 a, fvec4 b) {
  bf16x8 r;
#pragma unroll
  for (int i = 0; i < 4; ++i) { r[i] = (__bf16)a[i]; r[i + 4] = (__bf16)b[i]; }
  return r;
}

__device__ inline void hl8(fvec4 a, fvec4 b, bf16x8& h, bf16x8& l) {
#pragma unroll
  for (int i = 0; i < 4; ++i) {
    float x = a[i]; __bf16 hx = (__bf16)x; h[i] = hx; l[i] = (__bf16)(x - (float)hx);
    float y = b[i]; __bf16 hy = (__bf16)y; h[i + 4] = hy; l[i + 4] = (__bf16)(y - (float)hy);
  }
}

// ---- Wide fragment loaders --------------------------------------------------
// A (16x32): lane row = lane&15, kb=(lane>>4)*8; vector elems 0-7 <-> k=kb..kb+7,
//            elems 8-15 <-> k=kb+16..kb+23  (contiguous chunks)
// B (32x16): lane col = lane&15, kh=(lane>>4)*8; elems 0-7 <-> k=kh..kh+7, 8-15 <-> k=kh+16..kh+23.

__device__ inline bf16x16 load_a_g(const float* __restrict__ base, int ld, int row0,
                                   int k0, int lane) {
  const int r = lane & 15, kb = (lane >> 4) * 8;
  const float* p = base + (size_t)(row0 + r) * ld + k0 + kb;
  fvec4 x0 = *(const fvec4*)(p);
  fvec4 x1 = *(const fvec4*)(p + 4);
  fvec4 x2 = *(const fvec4*)(p + 16);
  fvec4 x3 = *(const fvec4*)(p + 20);
  return cat16(cvt8(x0, x1), cvt8(x2, x3));
}

__device__ inline void load_bT_hl_g(const float* __restrict__ base, int ld, int n0,
                                    int k0, int lane, bf16x16& hi, bf16x16& lo) {
  const int n = lane & 15, kh = (lane >> 4) * 8;
  const float* p = base + (size_t)(n0 + n) * ld + k0 + kh;
  fvec4 x0 = *(const fvec4*)(p);
  fvec4 x1 = *(const fvec4*)(p + 4);
  fvec4 x2 = *(const fvec4*)(p + 16);
  fvec4 x3 = *(const fvec4*)(p + 20);
  bf16x8 h0, l0, h1, l1;
  hl8(x0, x1, h0, l0);
  hl8(x2, x3, h1, l1);
  hi = cat16(h0, h1);
  lo = cat16(l0, l1);
}

__device__ inline bf16x16 load_bT_lds_f32(const float* __restrict__ base, int ld, int n0,
                                          int k0, int lane) {
  const int n = lane & 15, kh = (lane >> 4) * 8;
  const float* p = base + (size_t)(n0 + n) * ld + k0 + kh;
  fvec4 x0 = *(const fvec4*)(p);
  fvec4 x1 = *(const fvec4*)(p + 4);
  fvec4 x2 = *(const fvec4*)(p + 16);
  fvec4 x3 = *(const fvec4*)(p + 20);
  return cat16(cvt8(x0, x1), cvt8(x2, x3));
}

// LDS bf16, row stride 40 (80 B: 16-B aligned, 20-bank stride -> conflict free)
#define LDW 40

__device__ inline bf16x16 load_a_lds(const __bf16* __restrict__ base, int row0, int lane) {
  const int r = lane & 15, kb = (lane >> 4) * 8;
  const __bf16* p = base + (size_t)(row0 + r) * LDW + kb;
  return cat16(*(const bf16x8*)(p), *(const bf16x8*)(p + 16));
}

__device__ inline bf16x16 load_b_lds_k(const __bf16* __restrict__ base, int col, int lane) {
  const int kh = (lane >> 4) * 8;
  const __bf16* p = base + (size_t)col * LDW + kh;
  return cat16(*(const bf16x8*)(p), *(const bf16x8*)(p + 16));
}

// ---- Kernel 1: LePE conv as 3 shifted GEMMs; writes out = bias + conv -------
// Tile: 128(i) x 256(j) per WG; waves 2x4 -> 64x64 each.
__global__ __launch_bounds__(256) void lepe_kernel(const float* __restrict__ v,
                                                   const float* __restrict__ w,
                                                   const float* __restrict__ bias,
                                                   float* __restrict__ out) {
  __shared__ __align__(16) __bf16 Wl[3 * 128 * LDW];  // 30 KB
  __shared__ __align__(16) __bf16 Vt[258 * LDW];      // ~20.6 KB, transposed [col][k]

  const int jt = blockIdx.x, it = blockIdx.y, b = blockIdx.z;
  const int i0 = it * 128, j0 = jt * 256;
  const int t = threadIdx.x;
  const int lane = t & 31, wv = t >> 5;
  const int wi = wv >> 2, wj = wv & 3;

  const float* Vb = v + (size_t)b * SEQ * DIM;

  f32x8 acc[4][4];
#pragma unroll
  for (int mt = 0; mt < 4; ++mt)
#pragma unroll
    for (int nt = 0; nt < 4; ++nt) acc[mt][nt] = zero8();

  for (int c0 = 0; c0 < DIM; c0 += 32) {
    __syncthreads();
    // stage W chunk: Wl[d][r][cc] = conv_w[i0+r][c0+cc][1][d]   (L2-resident)
    for (int e = t; e < 3 * 128 * 32; e += 256) {
      const int d = e / (128 * 32);
      const int rem = e - d * 128 * 32;
      const int r = rem >> 5, cc = rem & 31;
      Wl[(d * 128 + r) * LDW + cc] =
          (__bf16)w[(size_t)(i0 + r) * (DIM * 9) + (size_t)(c0 + cc) * 9 + 3 + d];
    }
    // stage V chunk TRANSPOSED with halo: Vt[jj][cc] = V[c0+cc][j0-1+jj]
    for (int e = t; e < 258 * 32; e += 256) {
      const int cc = e / 258, jj = e - cc * 258;
      const int gj = j0 - 1 + jj;
      float x = (gj >= 0 && gj < DIM) ? Vb[(size_t)(c0 + cc) * DIM + gj] : 0.f;
      Vt[jj * LDW + cc] = (__bf16)x;
    }
    __syncthreads();

#pragma unroll
    for (int d = 0; d < 3; ++d) {
      bf16x16 afr[4];
#pragma unroll
      for (int mt = 0; mt < 4; ++mt)
        afr[mt] = load_a_lds(Wl + d * 128 * LDW, wi * 64 + mt * 16, lane);
#pragma unroll
      for (int nt = 0; nt < 4; ++nt) {
        const int col = wj * 64 + nt * 16 + (lane & 15) + d;
        bf16x16 bfr = load_b_lds_k(Vt, col, lane);
#pragma unroll
        for (int mt = 0; mt < 4; ++mt)
          acc[mt][nt] = wmma_bf16(afr[mt], bfr, acc[mt][nt]);
      }
    }
  }

  const int lh = lane >> 4, ln = lane & 15;
  float* ob = out + (size_t)b * SEQ * DIM;
#pragma unroll
  for (int mt = 0; mt < 4; ++mt)
#pragma unroll
    for (int nt = 0; nt < 4; ++nt)
#pragma unroll
      for (int g = 0; g < 8; ++g) {
        const int gi = i0 + wi * 64 + mt * 16 + lh * 8 + g;
        const int gj = j0 + wj * 64 + nt * 16 + ln;
        ob[(size_t)gi * DIM + gj] = acc[mt][nt][g] + bias[gi];
      }
}

// ---- Kernel 2: attention; accumulates V @ P^T into out ----------------------
#define QBLK 64
#define SLD  1028   // padded LDS row stride (floats); 4112 B keeps 16-B alignment
#define ATTN_SMEM_BYTES (((QBLK * SLD + QBLK * 4 + QBLK) * 4) + 2 * (QBLK * LDW * 2))

__global__ __launch_bounds__(256) void attn_kernel(const float* __restrict__ q,
                                                   const float* __restrict__ k,
                                                   const float* __restrict__ v,
                                                   float* __restrict__ out) {
  extern __shared__ __align__(16) char smem[];
  float* S       = (float*)smem;            // [QBLK][SLD] logits -> probabilities
  float* red     = S + QBLK * SLD;          // [QBLK][4]
  float* rowstat = red + QBLK * 4;          // [QBLK]
  __bf16* Qh     = (__bf16*)(rowstat + QBLK);  // [QBLK][LDW]
  __bf16* Ql     = Qh + QBLK * LDW;            // [QBLK][LDW]

  const int jb = blockIdx.x, b = blockIdx.y;
  const int j0 = jb * QBLK;
  const int t = threadIdx.x, lane = t & 31, wv = t >> 5;
  const int lh = lane >> 4, ln = lane & 15;

  const float* Qb = q + (size_t)b * SEQ * DIM;
  const float* Kb = k + (size_t)b * SEQ * DIM;
  const float* Vb = v + (size_t)b * SEQ * DIM;

  // ---- Phase 1: S[64][1024] = Q_blk @ K^T, fp32-faithful via bf16 hi/lo split
  for (int mc = 0; mc < 2; ++mc) {
    const int m0 = wv * 128 + mc * 64;
    f32x8 acc[4][4];
#pragma unroll
    for (int mt = 0; mt < 4; ++mt)
#pragma unroll
      for (int nt = 0; nt < 4; ++nt) acc[mt][nt] = zero8();

    for (int c0 = 0; c0 < DIM; c0 += 32) {
      __syncthreads();
      // cooperative hi/lo staging of the 64x32 Q chunk (wide loads, packed cvt)
#pragma unroll
      for (int s = 0; s < 2; ++s) {
        const int e = t + s * 256;
        const int row = e >> 3, col4 = (e & 7) * 4;
        fvec4 x = *(const fvec4*)(Qb + (size_t)(j0 + row) * DIM + c0 + col4);
        bf16x4 h4, l4;
#pragma unroll
        for (int i = 0; i < 4; ++i) {
          __bf16 hx = (__bf16)x[i];
          h4[i] = hx;
          l4[i] = (__bf16)(x[i] - (float)hx);
        }
        *(bf16x4*)(Qh + row * LDW + col4) = h4;
        *(bf16x4*)(Ql + row * LDW + col4) = l4;
      }
      __syncthreads();

      bf16x16 ah[4], al[4];
#pragma unroll
      for (int mt = 0; mt < 4; ++mt) {
        ah[mt] = load_a_lds(Qh, mt * 16, lane);
        al[mt] = load_a_lds(Ql, mt * 16, lane);
      }
#pragma unroll
      for (int nt = 0; nt < 4; ++nt) {
        // prefetch next K chunk for this lane's key row (global_prefetch_b8)
        if (c0 + 32 < DIM)
          __builtin_prefetch(Kb + (size_t)(m0 + nt * 16 + ln) * DIM + c0 + 32 + lh * 16, 0, 1);
        bf16x16 bh, bl;
        load_bT_hl_g(Kb, DIM, m0 + nt * 16, c0, lane, bh, bl);
#pragma unroll
        for (int mt = 0; mt < 4; ++mt) {
          acc[mt][nt] = wmma_bf16(ah[mt], bh, acc[mt][nt]);
          acc[mt][nt] = wmma_bf16(ah[mt], bl, acc[mt][nt]);
          acc[mt][nt] = wmma_bf16(al[mt], bh, acc[mt][nt]);
        }
      }
    }
#pragma unroll
    for (int mt = 0; mt < 4; ++mt)
#pragma unroll
      for (int nt = 0; nt < 4; ++nt)
#pragma unroll
        for (int g = 0; g < 8; ++g)
          S[(size_t)(mt * 16 + lh * 8 + g) * SLD + m0 + nt * 16 + ln] = acc[mt][nt][g];
  }
  __syncthreads();

  // ---- Softmax over each of the 64 rows (4 threads per row) ----
  {
    const int r = t >> 2, qd = t & 3;
    float* Sr = S + (size_t)r * SLD + qd * 256;
    float mx = -3.0e38f;
    for (int i = 0; i < 256; ++i) mx = fmaxf(mx, Sr[i]);
    red[r * 4 + qd] = mx;
    __syncthreads();
    if (qd == 0)
      rowstat[r] = fmaxf(fmaxf(red[r * 4], red[r * 4 + 1]),
                         fmaxf(red[r * 4 + 2], red[r * 4 + 3]));
    __syncthreads();
    const float m = rowstat[r];
    float s = 0.f;
    for (int i = 0; i < 256; ++i) {
      float e = __expf(Sr[i] - m);
      Sr[i] = e;
      s += e;
    }
    red[r * 4 + qd] = s;
    __syncthreads();
    if (qd == 0)
      rowstat[r] = 1.f / (red[r * 4] + red[r * 4 + 1] + red[r * 4 + 2] + red[r * 4 + 3]);
    __syncthreads();
    const float inv = rowstat[r];
    for (int i = 0; i < 256; ++i) Sr[i] *= inv;
  }
  __syncthreads();

  // ---- Phase 2: out[b, i, j0+j] += sum_m V[i,m] * P[j,m] ----
  // 4 i-tiles per iteration: 16 WMMA per k-step, P->bf16 conversion amortized 4x.
  float* ob = out + (size_t)b * SEQ * DIM;
  const int iw = wv * 128;
#pragma unroll 1
  for (int it = 0; it < 2; ++it) {
    const int i0 = iw + it * 64;
    f32x8 acc[4][4];  // [at][nt]
#pragma unroll
    for (int at = 0; at < 4; ++at)
#pragma unroll
      for (int nt = 0; nt < 4; ++nt) acc[at][nt] = zero8();

    for (int m0 = 0; m0 < SEQ; m0 += 32) {
      bf16x16 av[4];
#pragma unroll
      for (int at = 0; at < 4; ++at) {
        if (m0 + 32 < SEQ)
          __builtin_prefetch(Vb + (size_t)(i0 + at * 16 + ln) * DIM + m0 + 32 + lh * 8, 0, 1);
        av[at] = load_a_g(Vb, DIM, i0 + at * 16, m0, lane);
      }
#pragma unroll
      for (int nt = 0; nt < 4; ++nt) {
        bf16x16 bp = load_bT_lds_f32(S, SLD, nt * 16, m0, lane);
#pragma unroll
        for (int at = 0; at < 4; ++at)
          acc[at][nt] = wmma_bf16(av[at], bp, acc[at][nt]);
      }
    }
#pragma unroll
    for (int at = 0; at < 4; ++at)
#pragma unroll
      for (int nt = 0; nt < 4; ++nt)
#pragma unroll
        for (int g = 0; g < 8; ++g) {
          const int gi = i0 + at * 16 + lh * 8 + g;
          const int gj = j0 + nt * 16 + ln;
          ob[(size_t)gi * DIM + gj] += acc[at][nt][g];
        }
  }
}

// ---- Launch -----------------------------------------------------------------
extern "C" void kernel_launch(void* const* d_in, const int* in_sizes, int n_in,
                              void* d_out, int out_size, void* d_ws, size_t ws_size,
                              hipStream_t stream) {
  (void)in_sizes; (void)n_in; (void)d_ws; (void)ws_size; (void)out_size;
  const float* q  = (const float*)d_in[0];
  const float* k  = (const float*)d_in[1];
  const float* v  = (const float*)d_in[2];
  const float* cw = (const float*)d_in[3];
  const float* cb = (const float*)d_in[4];
  float* out = (float*)d_out;

  // 1) out = bias + conv (LePE)
  lepe_kernel<<<dim3(4, 8, BATCH), 256, 0, stream>>>(v, cw, cb, out);

  // 2) out += V @ softmax(QK^T)^T   (256KB+ dynamic LDS on the 320KB WGP)
  hipFuncSetAttribute(reinterpret_cast<const void*>(attn_kernel),
                      hipFuncAttributeMaxDynamicSharedMemorySize, ATTN_SMEM_BYTES);
  attn_kernel<<<dim3(SEQ / QBLK, BATCH), 256, ATTN_SMEM_BYTES, stream>>>(q, k, v, out);
}
